// ConformationVectorModule_63909113364923
// MI455X (gfx1250) — matrix operand order repaired and checked by reading, not compiled, not run on hardware
//
#include <hip/hip_runtime.h>
#include <hip/hip_bf16.h>

// ---------------------------------------------------------------------------
// CDNA5 / gfx1250 WMMA implementation of the ConformationVectorModule forward.
// wave32, v_wmma_f32_16x16x32_f16 for all matrix math,
// global_load_async_to_lds_b128 (ASYNCcnt) for the f16 weight-tile staging.
// ---------------------------------------------------------------------------

typedef _Float16 f16;
typedef __attribute__((ext_vector_type(16))) _Float16 v16h;
typedef __attribute__((ext_vector_type(8)))  float    v8f;

union AFragU {
    v16h v;
    uint4 q[2];
    _Float16 h[16];
};

union PkU {
    f16 h[4];
    uint2 u;
};

// Model constants
#define NSEQ   2048
#define CS     384
#define NHEAD  12
#define CHID   16
#define HC     192   // NHEAD*CHID
#define CRES   128
#define NBLK   8

static __device__ __forceinline__ v8f wmma_f16(v16h a, v16h b, v8f c) {
    return __builtin_amdgcn_wmma_f32_16x16x32_f16(false, a, false, b, (short)0, c,
                                                  false, false);
}

// Async copy 16 bytes global -> LDS (per lane), tracked with ASYNCcnt.
static __device__ __forceinline__ void async_copy_b128(void* lds_dst,
                                                       const void* gsrc) {
    unsigned lds_off = (unsigned)(uintptr_t)lds_dst;         // addr[31:0] = LDS offset
    unsigned long long ga = (unsigned long long)(uintptr_t)gsrc;
    asm volatile("global_load_async_to_lds_b128 %0, %1, off"
                 :: "v"(lds_off), "v"(ga)
                 : "memory");
}

static __device__ __forceinline__ void wait_async0() {
    asm volatile("s_wait_asynccnt 0" ::: "memory");
}

// ---------------------------------------------------------------------------
// Weight transpose + fp32->f16 convert: W[K][N] row-major -> Wt[N][K] f16
// ---------------------------------------------------------------------------
__global__ void wconv_kernel(const float* __restrict__ W, f16* __restrict__ Wt,
                             int K, int N) {
    int idx = blockIdx.x * 256 + threadIdx.x;
    if (idx >= K * N) return;
    int n = idx / K;
    int k = idx - n * K;
    Wt[idx] = (f16)W[(size_t)k * N + n];
}

// ---------------------------------------------------------------------------
// LayerNorm over last dim (384). One row per block of 384 threads.
// Safe when X == Y (each thread reads only its own element before writing).
// ---------------------------------------------------------------------------
__global__ __launch_bounds__(384) void ln384_kernel(const float* X, float* Y,
                                                    const float* __restrict__ g,
                                                    const float* __restrict__ b) {
    __shared__ float red[384];
    __shared__ float stat;
    int row = blockIdx.x, t = threadIdx.x;
    float x = X[(size_t)row * CS + t];
    red[t] = x;
    __syncthreads();
    for (int s = 192; s >= 6; s >>= 1) {
        if (t < s) red[t] += red[t + s];
        __syncthreads();
    }
    if (t == 0) {
        float m = 0.f;
        for (int i = 0; i < 6; ++i) m += red[i];
        stat = m * (1.f / CS);
    }
    __syncthreads();
    float mean = stat;
    float d = x - mean;
    __syncthreads();
    red[t] = d * d;
    __syncthreads();
    for (int s = 192; s >= 6; s >>= 1) {
        if (t < s) red[t] += red[t + s];
        __syncthreads();
    }
    if (t == 0) {
        float v = 0.f;
        for (int i = 0; i < 6; ++i) v += red[i];
        stat = v * (1.f / CS);
    }
    __syncthreads();
    float var = stat;
    Y[(size_t)row * CS + t] = d * rsqrtf(var + 1e-5f) * g[t] + b[t];
}

// ---------------------------------------------------------------------------
// Generic WMMA GEMM: C[M,N] = act(A[M,K]) @ Wt^T + bias (+ R), optional relus.
// Wt is [N][K] f16 (pre-transposed). 256 threads = 8 waves, tile 128x64,
// wave grid 4(M) x 2(N), each wave computes 32x32 = 2x2 WMMA 16x16 tiles.
// B tile staged by GLOBAL_LOAD_ASYNC_TO_LDS_B128 (one b128 per thread),
// A tile staged as float4 -> packed f16 -> ds_store_b64.
// Requires: M % 128 == 0, N % 64 == 0, K % 32 == 0.
// ---------------------------------------------------------------------------
__global__ __launch_bounds__(256) void gemm_f16_kernel(
    const float* __restrict__ A, const f16* __restrict__ Wt,
    const float* __restrict__ bias, const float* __restrict__ R,
    float* __restrict__ C, int K, int N, int reluIn, int reluOut) {
    __shared__ __align__(16) f16 As[128 * 32];
    __shared__ __align__(16) f16 Bs[64 * 32];

    int m0 = blockIdx.x * 128;
    int n0 = blockIdx.y * 64;
    int tid = threadIdx.x;
    int lane = tid & 31;
    int w = tid >> 5;
    int wm = w >> 1;       // 0..3
    int wn = w & 1;        // 0..1
    int hb = lane >> 4;    // lane half (0/1)
    int li = lane & 15;

    // Staging coordinates
    int br = tid >> 2;            // B: row (col of C), 0..63
    int bc = (tid & 3) * 8;       // B: k offset, one b128 each
    int ar = tid >> 3;            // A: row base, 0..31 (i adds 32 each)
    int ac = (tid & 7) * 4;       // A: k offset, one float4 each

    v8f acc[2][2] = {};

    for (int kb = 0; kb < K; kb += 32) {
        __syncthreads();
        // --- B tile: pure f16 copy, async to LDS ---
        async_copy_b128(&Bs[br * 32 + bc], &Wt[(size_t)(n0 + br) * K + kb + bc]);
        // --- A tile: float4 load -> 4 x f16 -> ds_store_b64 ---
#pragma unroll
        for (int i = 0; i < 4; ++i) {
            int row = i * 32 + ar;
            const float* gp = &A[(size_t)(m0 + row) * K + kb + ac];
            float4 v4 = *(const float4*)gp;
            if (reluIn) {
                v4.x = fmaxf(v4.x, 0.f);
                v4.y = fmaxf(v4.y, 0.f);
                v4.z = fmaxf(v4.z, 0.f);
                v4.w = fmaxf(v4.w, 0.f);
            }
            PkU pk;
            pk.h[0] = (f16)v4.x;
            pk.h[1] = (f16)v4.y;
            pk.h[2] = (f16)v4.z;
            pk.h[3] = (f16)v4.w;
            *(uint2*)&As[row * 32 + ac] = pk.u;
            if (kb + 32 < K) __builtin_prefetch(gp + 32, 0, 3);
        }
        wait_async0();
        __syncthreads();

        v16h afrag[2], bfrag[2];
#pragma unroll
        for (int i = 0; i < 2; ++i) {
            // A layout: row = li, halves 0..7 at K = hb*8, halves 8..15 at 16+hb*8
            const f16* base = &As[(wm * 32 + i * 16 + li) * 32 + hb * 8];
            AFragU u;
            u.q[0] = *(const uint4*)(base);
            u.q[1] = *(const uint4*)(base + 16);
            afrag[i] = u.v;
        }
#pragma unroll
        for (int j = 0; j < 2; ++j) {
            // B layout: col = li, halves 0..7 at K = hb*8, halves 8..15 at 16+hb*8
            const f16* base = &Bs[(wn * 32 + j * 16 + li) * 32 + hb * 8];
            AFragU u;
            u.q[0] = *(const uint4*)(base);
            u.q[1] = *(const uint4*)(base + 16);
            bfrag[j] = u.v;
        }
#pragma unroll
        for (int i = 0; i < 2; ++i)
#pragma unroll
            for (int j = 0; j < 2; ++j)
                acc[i][j] = wmma_f16(afrag[i], bfrag[j], acc[i][j]);
    }

    // Store: C/D layout -> M = r + 8*hb, N = li
#pragma unroll
    for (int i = 0; i < 2; ++i) {
#pragma unroll
        for (int j = 0; j < 2; ++j) {
#pragma unroll
            for (int r = 0; r < 8; ++r) {
                int row = m0 + wm * 32 + i * 16 + r + hb * 8;
                int col = n0 + wn * 32 + j * 16 + li;
                float v = acc[i][j][r] + bias[col];
                if (R) v += R[(size_t)row * N + col];
                if (reluOut) v = v > 0.f ? v : 0.f;
                C[(size_t)row * N + col] = v;
            }
        }
    }
}

// ---------------------------------------------------------------------------
// Flash attention. Grid: NHEAD * (NSEQ/128) blocks of 256 threads (8 waves).
// Each wave: 16 query rows of one head. Chunks of 32 keys.
// Q: [NSEQ][HC] fp32, KV: [NSEQ][2*HC] fp32 (k then v per head), O: [NSEQ][HC].
// No 1/sqrt(d) scale (reference has none); bias = 1e5*(mq*mk - 1).
// ---------------------------------------------------------------------------
__global__ __launch_bounds__(256) void attn_kernel(
    const float* __restrict__ Q, const float* __restrict__ KV,
    const float* __restrict__ mask, float* __restrict__ O) {
    __shared__ __align__(16) f16 Kt[32 * 16];    // [key][ch]
    __shared__ __align__(16) f16 Vt[16 * 32];    // [ch][key]
    __shared__ __align__(16) f16 Pt[8][16 * 32]; // per wave: [qrow][key]
    __shared__ float mk_s[32];

    int head = blockIdx.x / 16;
    int qt = blockIdx.x % 16;
    int tid = threadIdx.x;
    int lane = tid & 31;
    int w = tid >> 5;
    int hb = lane >> 4;
    int li = lane & 15;
    int qbase = qt * 128 + w * 16;

    // Q fragment: rows qbase+li, channels (hb*8 + h) for h<8, zeros for h>=8
    AFragU qu;
    {
        int qrow = qbase + li;
        const float* qp = &Q[(size_t)qrow * HC + head * CHID + hb * 8];
        float4 q0 = *(const float4*)(qp);
        float4 q1 = *(const float4*)(qp + 4);
        qu.h[0] = (f16)q0.x; qu.h[1] = (f16)q0.y;
        qu.h[2] = (f16)q0.z; qu.h[3] = (f16)q0.w;
        qu.h[4] = (f16)q1.x; qu.h[5] = (f16)q1.y;
        qu.h[6] = (f16)q1.z; qu.h[7] = (f16)q1.w;
#pragma unroll
        for (int h = 8; h < 16; ++h) qu.h[h] = (f16)0.f;
    }
    v16h qfrag = qu.v;

    float mq[8];
#pragma unroll
    for (int r = 0; r < 8; ++r) mq[r] = mask[qbase + r + hb * 8];

    float mstat[8], lstat[8];
#pragma unroll
    for (int r = 0; r < 8; ++r) { mstat[r] = -1e30f; lstat[r] = 0.f; }
    v8f oacc = {};

    for (int kc = 0; kc < NSEQ; kc += 32) {
        __syncthreads();
        // Stage K chunk: Kt[key][ch], float4 -> packed f16 (threads 0..127)
        if (tid < 128) {
            int key = tid >> 2, ch = (tid & 3) * 4;
            float4 k4 = *(const float4*)&KV[(size_t)(kc + key) * (2 * HC) +
                                            head * 32 + ch];
            PkU pk;
            pk.h[0] = (f16)k4.x; pk.h[1] = (f16)k4.y;
            pk.h[2] = (f16)k4.z; pk.h[3] = (f16)k4.w;
            *(uint2*)&Kt[key * 16 + ch] = pk.u;
        }
        // Stage V chunk transposed: Vt[ch][key] (2 scalar elements / thread)
#pragma unroll
        for (int i = 0; i < 2; ++i) {
            int e = tid + i * 256;
            int ch = e >> 5, key = e & 31;
            Vt[e] = (f16)KV[(size_t)(kc + key) * (2 * HC) + head * 32 + 16 + ch];
        }
        if (tid < 32) mk_s[tid] = mask[kc + tid];
        __syncthreads();

        // K fragments as B matrices. Contraction dim = channel (padded to 32):
        // elements 0..7 hold channels hb*8..hb*8+7; elements 8..15 hold zeros.
        v16h b0, b1;
        {
            AFragU u0, u1;
            const f16* p0 = &Kt[li * 16 + hb * 8];
            u0.q[0] = *(const uint4*)p0;
            u0.q[1] = make_uint4(0u, 0u, 0u, 0u);
            const f16* p1 = &Kt[(16 + li) * 16 + hb * 8];
            u1.q[0] = *(const uint4*)p1;
            u1.q[1] = make_uint4(0u, 0u, 0u, 0u);
            b0 = u0.v; b1 = u1.v;
        }

        v8f S0 = {}, S1 = {};
        S0 = wmma_f16(qfrag, b0, S0);
        S1 = wmma_f16(qfrag, b1, S1);

        float mk0 = mk_s[li], mk1 = mk_s[16 + li];
        float P0[8], P1[8];
#pragma unroll
        for (int r = 0; r < 8; ++r) {
            float s0 = S0[r] + 100000.f * (mq[r] * mk0 - 1.f);
            float s1 = S1[r] + 100000.f * (mq[r] * mk1 - 1.f);
            float rm = fmaxf(s0, s1);
#pragma unroll
            for (int m2 = 8; m2 >= 1; m2 >>= 1)
                rm = fmaxf(rm, __shfl_xor(rm, m2, 32));
            float mn = fmaxf(mstat[r], rm);
            float sc = __expf(mstat[r] - mn);
            float p0 = __expf(s0 - mn), p1 = __expf(s1 - mn);
            float rs = p0 + p1;
#pragma unroll
            for (int m2 = 8; m2 >= 1; m2 >>= 1)
                rs += __shfl_xor(rs, m2, 32);
            lstat[r] = lstat[r] * sc + rs;
            mstat[r] = mn;
            oacc[r] *= sc;
            P0[r] = p0;
            P1[r] = p1;
        }

        // Round-trip P through per-wave LDS patch into A-fragment layout.
        f16* pw = Pt[w];
#pragma unroll
        for (int r = 0; r < 8; ++r) {
            int prow = r + hb * 8;
            pw[prow * 32 + li] = (f16)P0[r];
            pw[prow * 32 + 16 + li] = (f16)P1[r];
        }
        asm volatile("s_wait_dscnt 0" ::: "memory");

        AFragU pu;
        {
            const f16* pb = &pw[li * 32 + hb * 8];
            pu.q[0] = *(const uint4*)pb;
            pu.q[1] = *(const uint4*)(pb + 16);
        }
        AFragU vu;
        {
            const f16* vb = &Vt[li * 32 + hb * 8];
            vu.q[0] = *(const uint4*)vb;
            vu.q[1] = *(const uint4*)(vb + 16);
        }
        oacc = wmma_f16(pu.v, vu.v, oacc);
    }

    // Normalize and write O
#pragma unroll
    for (int r = 0; r < 8; ++r) {
        int row = qbase + r + hb * 8;
        O[(size_t)row * HC + head * CHID + li] = oacc[r] / lstat[r];
    }
}

// ---------------------------------------------------------------------------
// Resnet head: a -> relu(a) -> theta/phi/r + normalization, write to d_out.
// 128 threads = 4 waves, one row per wave. 2048/4 = 512 blocks.
// ---------------------------------------------------------------------------
__global__ __launch_bounds__(128) void head_kernel(
    const float* __restrict__ A, const float* __restrict__ Wtheta,
    const float* __restrict__ btheta, const float* __restrict__ Wphi,
    const float* __restrict__ bphi, const float* __restrict__ Wr,
    const float* __restrict__ br, float* __restrict__ out, int bi) {
    int w = threadIdx.x >> 5, lane = threadIdx.x & 31;
    int row = blockIdx.x * 4 + w;
    const float* a = A + (size_t)row * CRES;

    float4 a4 = *(const float4*)&a[lane * 4];
    float tv[4] = {fmaxf(a4.x, 0.f), fmaxf(a4.y, 0.f), fmaxf(a4.z, 0.f),
                   fmaxf(a4.w, 0.f)};

    float p[5] = {0.f, 0.f, 0.f, 0.f, 0.f};
#pragma unroll
    for (int i = 0; i < 4; ++i) {
        int k = lane * 4 + i;
        p[0] += tv[i] * Wtheta[k * 2 + 0];
        p[1] += tv[i] * Wtheta[k * 2 + 1];
        p[2] += tv[i] * Wphi[k * 2 + 0];
        p[3] += tv[i] * Wphi[k * 2 + 1];
        p[4] += tv[i] * Wr[k];
    }
#pragma unroll
    for (int m = 16; m >= 1; m >>= 1)
#pragma unroll
        for (int i = 0; i < 5; ++i) p[i] += __shfl_xor(p[i], m, 32);

    if (lane == 0) {
        float th0 = p[0] + btheta[0], th1 = p[1] + btheta[1];
        float ph0 = p[2] + bphi[0], ph1 = p[3] + bphi[1];
        ph1 = ph1 > 0.f ? ph1 : 0.f;  // relu on phi[...,1] before norm
        float rr = p[4] + br[0];
        float tn = rsqrtf(fmaxf(th0 * th0 + th1 * th1, 1e-8f));
        float pn = rsqrtf(fmaxf(ph0 * ph0 + ph1 * ph1, 1e-8f));
        // Output sections (floats): un_theta@0, theta@32768, un_phi@65536,
        // phi@98304, r@131072, s@147456
        size_t b2 = ((size_t)bi * NSEQ + row) * 2;
        out[b2] = th0;              out[b2 + 1] = th1;
        out[32768 + b2] = th0 * tn; out[32768 + b2 + 1] = th1 * tn;
        out[65536 + b2] = ph0;      out[65536 + b2 + 1] = ph1;
        out[98304 + b2] = ph0 * pn; out[98304 + b2 + 1] = ph1 * pn;
        out[131072 + (size_t)bi * NSEQ + row] = rr;
    }
}

// ---------------------------------------------------------------------------
// Host-side orchestration
// ---------------------------------------------------------------------------
extern "C" void kernel_launch(void* const* d_in, const int* in_sizes, int n_in,
                              void* d_out, int out_size, void* d_ws, size_t ws_size,
                              hipStream_t stream) {
    (void)in_sizes; (void)n_in; (void)out_size; (void)ws_size;

    // Input order: jax pytree flattening (sorted dict keys at every level).
    const float* mask   = (const float*)d_in[0];
    const float* Wkv    = (const float*)d_in[1];
    const float* bkv    = (const float*)d_in[2];
    const float* Wout   = (const float*)d_in[3];
    const float* bout   = (const float*)d_in[4];
    const float* Wq     = (const float*)d_in[5];
    const float* bq     = (const float*)d_in[6];
    const float* Wlin   = (const float*)d_in[7];
    const float* blin   = (const float*)d_in[8];
    const float* g_attn = (const float*)d_in[9];
    const float* b_attn = (const float*)d_in[10];
    const float* g_s    = (const float*)d_in[11];
    const float* b_s    = (const float*)d_in[12];
    const float* Wphi   = (const float*)d_in[13];
    const float* bphi   = (const float*)d_in[14];
    const float* Wr     = (const float*)d_in[15];
    const float* br     = (const float*)d_in[16];
    const float* Wb0l1  = (const float*)d_in[17];
    const float* bb0l1  = (const float*)d_in[18];
    const float* Wb0l2  = (const float*)d_in[19];
    const float* bb0l2  = (const float*)d_in[20];
    const float* Wb1l1  = (const float*)d_in[21];
    const float* bb1l1  = (const float*)d_in[22];
    const float* Wb1l2  = (const float*)d_in[23];
    const float* bb1l2  = (const float*)d_in[24];
    const float* Wri    = (const float*)d_in[25];
    const float* bri    = (const float*)d_in[26];
    const float* Wrinit = (const float*)d_in[27];
    const float* brinit = (const float*)d_in[28];
    const float* Wtheta = (const float*)d_in[29];
    const float* btheta = (const float*)d_in[30];
    const float* Wl1    = (const float*)d_in[31];
    const float* bl1    = (const float*)d_in[32];
    const float* Wl2    = (const float*)d_in[33];
    const float* bl2    = (const float*)d_in[34];
    const float* Wl3    = (const float*)d_in[35];
    const float* bl3    = (const float*)d_in[36];
    const float* g_t    = (const float*)d_in[37];
    const float* b_t    = (const float*)d_in[38];
    const float* single = (const float*)d_in[39];

    // Workspace carve-up
    float* f = (float*)d_ws;
    float* sA    = f; f += NSEQ * CS;
    float* sB    = f; f += NSEQ * CS;
    float* qb    = f; f += NSEQ * HC;
    float* kvb   = f; f += NSEQ * 2 * HC;
    float* ob    = f; f += NSEQ * HC;
    float* t1b   = f; f += NSEQ * CS;
    float* t2b   = f; f += NSEQ * CS;
    float* aInit = f; f += NSEQ * CRES;
    float* aA    = f; f += NSEQ * CRES;
    float* aB    = f; f += NSEQ * CRES;
    float* uB    = f; f += NSEQ * CRES;
    f16* h = (f16*)f;
    f16* Wq_t    = h; h += CS * HC;
    f16* Wkv_t   = h; h += CS * 2 * HC;
    f16* Wout_t  = h; h += HC * CS;
    f16* Wlin_t  = h; h += CS * CS;
    f16* Wl1_t   = h; h += CS * CS;
    f16* Wl2_t   = h; h += CS * CS;
    f16* Wl3_t   = h; h += CS * CS;
    f16* Wri_t   = h; h += CS * CRES;
    f16* Wrin_t  = h; h += CS * CRES;
    f16* Wb0l1_t = h; h += CRES * CRES;
    f16* Wb0l2_t = h; h += CRES * CRES;
    f16* Wb1l1_t = h; h += CRES * CRES;
    f16* Wb1l2_t = h; h += CRES * CRES;

    auto WC = [&](const float* W, f16* Wt, int K, int N) {
        int total = K * N;
        wconv_kernel<<<(total + 255) / 256, 256, 0, stream>>>(W, Wt, K, N);
    };
    auto G = [&](const float* A, const f16* Wt, const float* bias, const float* R,
                 float* C, int K, int N, int reluIn, int reluOut) {
        gemm_f16_kernel<<<dim3(NSEQ / 128, N / 64), 256, 0, stream>>>(
            A, Wt, bias, R, C, K, N, reluIn, reluOut);
    };
    auto LN = [&](const float* X, float* Y, const float* g, const float* b) {
        ln384_kernel<<<NSEQ, CS, 0, stream>>>(X, Y, g, b);
    };

    // --- Prep ---
    LN(single, sA, g_s, b_s);                 // s_initial
    WC(Wq, Wq_t, CS, HC);
    WC(Wkv, Wkv_t, CS, 2 * HC);
    WC(Wout, Wout_t, HC, CS);
    WC(Wlin, Wlin_t, CS, CS);
    WC(Wl1, Wl1_t, CS, CS);
    WC(Wl2, Wl2_t, CS, CS);
    WC(Wl3, Wl3_t, CS, CS);
    WC(Wri, Wri_t, CS, CRES);
    WC(Wrinit, Wrin_t, CS, CRES);
    WC(Wb0l1, Wb0l1_t, CRES, CRES);
    WC(Wb0l2, Wb0l2_t, CRES, CRES);
    WC(Wb1l1, Wb1l1_t, CRES, CRES);
    WC(Wb1l2, Wb1l2_t, CRES, CRES);

    // Block-invariant: aInit = relu(s_initial) @ res_init + b_init
    G(sA, Wrin_t, brinit, nullptr, aInit, CS, CRES, 1, 0);
    // s = s_initial @ lin_in + b
    G(sA, Wlin_t, blin, nullptr, sB, CS, CS, 0, 0);
    float* cur = sB;
    float* alt = sA;

    for (int bi = 0; bi < NBLK; ++bi) {
        // Attention
        G(cur, Wq_t, bq, nullptr, qb, CS, HC, 0, 0);
        G(cur, Wkv_t, bkv, nullptr, kvb, CS, 2 * HC, 0, 0);
        attn_kernel<<<NHEAD * (NSEQ / 128), 256, 0, stream>>>(qb, kvb, mask, ob);
        G(ob, Wout_t, bout, cur, alt, HC, CS, 0, 0);  // s = s + attn_out
        { float* tmp = cur; cur = alt; alt = tmp; }
        LN(cur, cur, g_attn, b_attn);

        // Transition
        G(cur, Wl1_t, bl1, nullptr, t1b, CS, CS, 0, 1);
        G(t1b, Wl2_t, bl2, nullptr, t2b, CS, CS, 0, 1);
        G(t2b, Wl3_t, bl3, cur, alt, CS, CS, 0, 0);   // s = s + t@l3
        { float* tmp = cur; cur = alt; alt = tmp; }
        LN(cur, cur, g_t, b_t);

        // Resnet trunk
        G(cur, Wri_t, bri, aInit, aA, CS, CRES, 1, 0);
        G(aA, Wb0l1_t, bb0l1, nullptr, uB, CRES, CRES, 1, 0);
        G(uB, Wb0l2_t, bb0l2, aA, aB, CRES, CRES, 1, 0);
        G(aB, Wb1l1_t, bb1l1, nullptr, uB, CRES, CRES, 1, 0);
        G(uB, Wb1l2_t, bb1l2, aB, aA, CRES, CRES, 1, 0);

        // Heads -> d_out sections
        head_kernel<<<NSEQ / 4, 128, 0, stream>>>(aA, Wtheta, btheta, Wphi, bphi,
                                                  Wr, br, (float*)d_out, bi);
    }

    // Final s -> d_out tail section
    hipMemcpyAsync((float*)d_out + 147456, cur, (size_t)NSEQ * CS * sizeof(float),
                   hipMemcpyDeviceToDevice, stream);
}
